// _PTDeformableFeatureAggregation_22411139350620
// MI455X (gfx1250) — matrix operand order repaired and checked by reading, not compiled, not run on hardware
//
#include <hip/hip_runtime.h>
#include <math.h>

// Problem constants (from reference)
#define E_    256
#define G_    8
#define GD_   32
#define NL_   4
#define NC_   6
#define NPTS_ 13
#define NLP_  6
#define BS_   2
#define NA_   900
#define WOUT_ 416   // G*NL*NPTS
#define NROWS_W (BS_*NA_*NC_)   // 10800
#define NROWS_O (BS_*NA_)       // 1800

typedef __attribute__((ext_vector_type(16))) _Float16 v16h;
typedef __attribute__((ext_vector_type(8)))  _Float16 v8h;
typedef __attribute__((ext_vector_type(8)))  float    v8f;

__device__ __constant__ float c_fix_scale[7][3] = {
    {0.f,0.f,0.f},{0.45f,0.f,0.f},{-0.45f,0.f,0.f},{0.f,0.45f,0.f},
    {0.f,-0.45f,0.f},{0.f,0.f,0.45f},{0.f,0.f,-0.45f}};
__device__ __constant__ int c_fm_h[4] = {64,32,16,8};
__device__ __constant__ int c_fm_w[4] = {176,88,44,22};

// ---------------------------------------------------------------------------
// K0: weight pre-pass: dst[n*K + k] = (f16) src[k*N + n]
// (N-major f16 panel so B fragments become contiguous 32-byte loads)
// ---------------------------------------------------------------------------
__global__ void k_transpose_f16(const float* __restrict__ src,
                                _Float16* __restrict__ dst, int K, int N) {
  const int i = blockIdx.x * blockDim.x + threadIdx.x;
  if (i < K * N) {
    const int n = i / K, k = i % K;           // consecutive threads: same n, k++
    dst[(size_t)n * K + k] = (_Float16)src[(size_t)k * N + n];
  }
}

// ---------------------------------------------------------------------------
// K1: camera embedding (12 rows). 1 block, 256 threads; thread = channel.
// ---------------------------------------------------------------------------
__global__ void k_cam_embed(const float* __restrict__ proj,
                            const float* __restrict__ w1, const float* __restrict__ b1,
                            const float* __restrict__ g1, const float* __restrict__ bb1,
                            const float* __restrict__ w2, const float* __restrict__ b2,
                            const float* __restrict__ g2, const float* __restrict__ bb2,
                            float* __restrict__ ce) {
  __shared__ float sh[E_];
  __shared__ float red[E_];
  const int e = threadIdx.x;
  for (int r = 0; r < BS_*NC_; ++r) {
    const float* cin = proj + r * 16;         // first 12 = rows 0..2 of 4x4
    float acc = b1[e];
#pragma unroll
    for (int j = 0; j < 12; ++j) acc += cin[j] * w1[j * E_ + e];
    acc = fmaxf(acc, 0.f);
    red[e] = acc; __syncthreads();
    for (int s = 128; s > 0; s >>= 1) { if (e < s) red[e] += red[e + s]; __syncthreads(); }
    float mean = red[0] * (1.f / E_); __syncthreads();
    float d = acc - mean;
    red[e] = d * d; __syncthreads();
    for (int s = 128; s > 0; s >>= 1) { if (e < s) red[e] += red[e + s]; __syncthreads(); }
    float var = red[0] * (1.f / E_); __syncthreads();
    float h1 = d * rsqrtf(var + 1e-5f) * g1[e] + bb1[e];
    sh[e] = h1; __syncthreads();
    float acc2 = b2[e];
    for (int k = 0; k < E_; ++k) acc2 += sh[k] * w2[k * E_ + e];
    acc2 = fmaxf(acc2, 0.f);
    __syncthreads();
    red[e] = acc2; __syncthreads();
    for (int s = 128; s > 0; s >>= 1) { if (e < s) red[e] += red[e + s]; __syncthreads(); }
    float mean2 = red[0] * (1.f / E_); __syncthreads();
    float d2 = acc2 - mean2;
    red[e] = d2 * d2; __syncthreads();
    for (int s = 128; s > 0; s >>= 1) { if (e < s) red[e] += red[e + s]; __syncthreads(); }
    float var2 = red[0] * (1.f / E_); __syncthreads();
    ce[r * E_ + e] = d2 * rsqrtf(var2 + 1e-5f) * g2[e] + bb2[e];
  }
}

// ---------------------------------------------------------------------------
// K2: keypoints + projection -> grid coords.  thread = anchor (1800 total)
// grid layout: [bs*NC, na, NPTS, 2]
// ---------------------------------------------------------------------------
__global__ void k_keypoints(const float* __restrict__ inst,
                            const float* __restrict__ anchor,
                            const float* __restrict__ lfc_w,
                            const float* __restrict__ lfc_b,
                            const float* __restrict__ proj,
                            const float* __restrict__ image_wh,
                            float* __restrict__ grid) {
  const int t = blockIdx.x * blockDim.x + threadIdx.x;
  if (t >= BS_ * NA_) return;
  const int b = t / NA_, n = t % NA_;
  const float* a = anchor + (size_t)t * 11;
  const float cx = a[0], cy = a[1], cz = a[2];
  const float sw = expf(a[3]), sl = expf(a[4]), shh = expf(a[5]);
  const float sy = a[6], cyaw = a[7];

  const float* f = inst + (size_t)t * E_;
  float accj[NLP_ * 3];
#pragma unroll
  for (int j = 0; j < NLP_ * 3; ++j) accj[j] = lfc_b[j];
  for (int k = 0; k < E_; ++k) {
    const float fv = f[k];
    const float* wr = lfc_w + (size_t)k * (NLP_ * 3);
#pragma unroll
    for (int j = 0; j < NLP_ * 3; ++j) accj[j] += fv * wr[j];
  }

  float kp[NPTS_][3];
#pragma unroll
  for (int i = 0; i < 7; ++i) {
    kp[i][0] = c_fix_scale[i][0] * sw;
    kp[i][1] = c_fix_scale[i][1] * sl;
    kp[i][2] = c_fix_scale[i][2] * shh;
  }
#pragma unroll
  for (int j = 0; j < NLP_ * 3; ++j) {
    const float s = 1.f / (1.f + expf(-accj[j])) - 0.5f;
    const int p = 7 + j / 3, d = j % 3;
    const float sz = (d == 0) ? sw : ((d == 1) ? sl : shh);
    kp[p][d] = s * sz;
  }

  float key[NPTS_][3];
#pragma unroll
  for (int p = 0; p < NPTS_; ++p) {
    key[p][0] = cyaw * kp[p][0] - sy * kp[p][1] + cx;
    key[p][1] = sy   * kp[p][0] + cyaw * kp[p][1] + cy;
    key[p][2] = kp[p][2] + cz;
  }

  for (int c = 0; c < NC_; ++c) {
    const float* P = proj + (size_t)(b * NC_ + c) * 16;
    const float whx = image_wh[(b * NC_ + c) * 2 + 0];
    const float why = image_wh[(b * NC_ + c) * 2 + 1];
#pragma unroll
    for (int p = 0; p < NPTS_; ++p) {
      const float X = key[p][0], Y = key[p][1], Z = key[p][2];
      const float u = P[0] * X + P[1] * Y + P[2]  * Z + P[3];
      const float v = P[4] * X + P[5] * Y + P[6]  * Z + P[7];
      float       z = P[8] * X + P[9] * Y + P[10] * Z + P[11];
      z = fmaxf(z, 1e-5f);
      const float gx = (u / z) / whx * 2.f - 1.f;
      const float gy = (v / z) / why * 2.f - 1.f;
      const size_t gi = (((size_t)(b * NC_ + c) * NA_ + n) * NPTS_ + p) * 2;
      grid[gi + 0] = gx;
      grid[gi + 1] = gy;
    }
  }
}

// ---------------------------------------------------------------------------
// WMMA GEMM core (contiguous k order, same for A and B):
//   A 16x32 fragment, lane L: M=L%16, half=L/16
//     -> elements = As[M][kbase+half*16 .. +15]
//   B 32x16 fragment, lane L: N=L%16, half=L/16
//     -> elements = Bt[N][kbase+half*16 .. +15]  (N-major f16 panel)
// 32-row M tile per block; 4 waves; wave owns one 16-wide N tile x 2 M tiles.
// ---------------------------------------------------------------------------
__device__ __forceinline__ v16h load_a_frag(const _Float16* As, int M, int kbase,
                                            int hf, int lds_stride) {
  const v8h lo = *(const v8h*)(As + (size_t)M * lds_stride + kbase + hf * 16);
  const v8h hi = *(const v8h*)(As + (size_t)M * lds_stride + kbase + hf * 16 + 8);
  return __builtin_shufflevector(lo, hi, 0,1,2,3,4,5,6,7,8,9,10,11,12,13,14,15);
}

// K3: w = (inst + aemb + ce) @ wfc_w + wfc_b   rows=10800, K=256, N=416
__global__ void __launch_bounds__(128) k_wfc_gemm(
    const float* __restrict__ inst, const float* __restrict__ aemb,
    const float* __restrict__ ce,
    const _Float16* __restrict__ Bt,      // [416][256] f16, N-major
    const float* __restrict__ bias,
    float* __restrict__ wout) {
  __shared__ _Float16 As[32 * 264];
  const int tid = threadIdx.x;
  const int rowbase = blockIdx.x * 32;

  for (int i = tid; i < 32 * E_; i += 128) {
    const int r = i >> 8, k = i & 255;
    const int gr = rowbase + r;
    float v = 0.f;
    if (gr < NROWS_W) {
      const int b = gr / (NA_ * NC_);
      const int rem = gr % (NA_ * NC_);
      const int n = rem / NC_, c = rem % NC_;
      v = inst[(size_t)(b * NA_ + n) * E_ + k]
        + aemb[(size_t)(b * NA_ + n) * E_ + k]
        + ce[(b * NC_ + c) * E_ + k];
    }
    As[r * 264 + k] = (_Float16)v;
  }
  __syncthreads();

  const int wave = tid >> 5, lane = tid & 31;
  const int ntile = blockIdx.y * 4 + wave;        // wave-uniform
  if (ntile >= WOUT_ / 16) return;                // EXEC stays all-ones per wave
  const int M = lane & 15, hf = lane >> 4;
  const int ncol = ntile * 16 + (lane & 15);
  const _Float16* bp = Bt + (size_t)ncol * E_ + hf * 16;

  v8f acc0 = {}, acc1 = {};
#pragma unroll
  for (int ks = 0; ks < 8; ++ks) {
    const int kbase = ks * 32;
    const v16h bf = *(const v16h*)(bp + kbase);               // 32B contiguous
    const v16h a0 = load_a_frag(As, M,      kbase, hf, 264);
    const v16h a1 = load_a_frag(As, M + 16, kbase, hf, 264);
    acc0 = __builtin_amdgcn_wmma_f32_16x16x32_f16(false, a0, false, bf,
                                                  (short)0, acc0, false, false);
    acc1 = __builtin_amdgcn_wmma_f32_16x16x32_f16(false, a1, false, bf,
                                                  (short)0, acc1, false, false);
  }
  const float bv = bias[ncol];
#pragma unroll
  for (int v = 0; v < 8; ++v) {
    const int r0 = rowbase + v + hf * 8;
    const int r1 = r0 + 16;
    if (r0 < NROWS_W) wout[(size_t)r0 * WOUT_ + ncol] = acc0[v] + bv;
    if (r1 < NROWS_W) wout[(size_t)r1 * WOUT_ + ncol] = acc1[v] + bv;
  }
}

// ---------------------------------------------------------------------------
// K4: softmax over (cam,level,point)=312 per (anchor, group), in place.
// ---------------------------------------------------------------------------
__global__ void k_softmax(float* __restrict__ w) {
  const int bn = blockIdx.x;
  const int g = threadIdx.x >> 5;
  const int lane = threadIdx.x & 31;
  float* base = w + (size_t)bn * NC_ * WOUT_;
  const int total = NC_ * NL_ * NPTS_;   // 312

  float lv[10];
  int cnt = 0;
  float m = -1e30f;
  for (int idx = lane; idx < total; idx += 32) {
    const int c = idx / (NL_ * NPTS_);
    const int r = idx % (NL_ * NPTS_);
    const float v = base[c * WOUT_ + r * G_ + g];
    lv[cnt++] = v;
    m = fmaxf(m, v);
  }
#pragma unroll
  for (int off = 16; off; off >>= 1) m = fmaxf(m, __shfl_xor(m, off, 32));
  float s = 0.f;
  for (int i = 0; i < cnt; ++i) { lv[i] = expf(lv[i] - m); s += lv[i]; }
#pragma unroll
  for (int off = 16; off; off >>= 1) s += __shfl_xor(s, off, 32);
  const float inv = 1.f / s;
  cnt = 0;
  for (int idx = lane; idx < total; idx += 32) {
    const int c = idx / (NL_ * NPTS_);
    const int r = idx % (NL_ * NPTS_);
    base[c * WOUT_ + r * G_ + g] = lv[cnt++] * inv;
  }
}

// ---------------------------------------------------------------------------
// K5: bilinear gather + weighted fusion.  block = anchor, thread = channel.
// ---------------------------------------------------------------------------
__global__ void k_aggregate(const float* __restrict__ fm0, const float* __restrict__ fm1,
                            const float* __restrict__ fm2, const float* __restrict__ fm3,
                            const float* __restrict__ grid, const float* __restrict__ w,
                            float* __restrict__ fused) {
  const int bn = blockIdx.x;
  const int b = bn / NA_, n = bn % NA_;
  const int e = threadIdx.x;
  const int g = e >> 5;
  const float* fml[4] = {fm0, fm1, fm2, fm3};
  const float* wrow = w + (size_t)bn * NC_ * WOUT_;
  float acc = 0.f;

  for (int c = 0; c < NC_; ++c) {
    const float* gp = grid + (((size_t)(b * NC_ + c) * NA_ + n) * NPTS_) * 2;
    const float* wc = wrow + c * WOUT_;
    __builtin_prefetch(wc + g, 0, 1);
    // per-(c,l) channel base pointers hoisted out of the point loop
    const float* fb_l[NL_];
#pragma unroll
    for (int l = 0; l < NL_; ++l)
      fb_l[l] = fml[l] + (size_t)((b * NC_ + c) * E_ + e) * (size_t)(c_fm_h[l] * c_fm_w[l]);
    for (int p = 0; p < NPTS_; ++p) {
      const float gx = gp[p * 2 + 0];
      const float gy = gp[p * 2 + 1];
#pragma unroll
      for (int l = 0; l < NL_; ++l) {
        const int H = c_fm_h[l], W = c_fm_w[l];
        const float px = (gx + 1.f) * (W * 0.5f) - 0.5f;
        const float py = (gy + 1.f) * (H * 0.5f) - 0.5f;
        const float x0f = floorf(px), y0f = floorf(py);
        const int x0 = (int)x0f, y0 = (int)y0f;
        const float wx1 = px - x0f, wy1 = py - y0f;
        const float wx0 = 1.f - wx1, wy0 = 1.f - wy1;
        const float* fb = fb_l[l];
        float v00 = 0.f, v01 = 0.f, v10 = 0.f, v11 = 0.f;
        const bool xi0 = (unsigned)x0 < (unsigned)W;
        const bool xi1 = (unsigned)(x0 + 1) < (unsigned)W;
        const bool yi0 = (unsigned)y0 < (unsigned)H;
        const bool yi1 = (unsigned)(y0 + 1) < (unsigned)H;
        if (xi0 && yi0) v00 = fb[y0 * W + x0];
        if (xi1 && yi0) v01 = fb[y0 * W + x0 + 1];
        if (xi0 && yi1) v10 = fb[(y0 + 1) * W + x0];
        if (xi1 && yi1) v11 = fb[(y0 + 1) * W + x0 + 1];
        const float samp = v00 * (wx0 * wy0) + v01 * (wx1 * wy0)
                         + v10 * (wx0 * wy1) + v11 * (wx1 * wy1);
        acc += samp * wc[(l * NPTS_ + p) * G_ + g];
      }
    }
  }
  fused[(size_t)bn * E_ + e] = acc;
}

// ---------------------------------------------------------------------------
// K6: out[:, :256] = fused @ op_w + op_b   rows=1800, K=256, N=256 (WMMA)
// ---------------------------------------------------------------------------
__global__ void __launch_bounds__(128) k_out_gemm(
    const float* __restrict__ fused,
    const _Float16* __restrict__ Bt,      // [256][256] f16, N-major
    const float* __restrict__ bias,
    float* __restrict__ out) {
  __shared__ _Float16 As[32 * 264];
  const int tid = threadIdx.x;
  const int rowbase = blockIdx.x * 32;

  for (int i = tid; i < 32 * E_; i += 128) {
    const int r = i >> 8, k = i & 255;
    const int gr = rowbase + r;
    const float v = (gr < NROWS_O) ? fused[(size_t)gr * E_ + k] : 0.f;
    As[r * 264 + k] = (_Float16)v;
  }
  __syncthreads();

  const int wave = tid >> 5, lane = tid & 31;
  const int ntile = blockIdx.y * 4 + wave;   // 0..15, always valid
  const int M = lane & 15, hf = lane >> 4;
  const int ncol = ntile * 16 + (lane & 15);
  const _Float16* bp = Bt + (size_t)ncol * E_ + hf * 16;

  v8f acc0 = {}, acc1 = {};
#pragma unroll
  for (int ks = 0; ks < 8; ++ks) {
    const int kbase = ks * 32;
    const v16h bf = *(const v16h*)(bp + kbase);
    const v16h a0 = load_a_frag(As, M,      kbase, hf, 264);
    const v16h a1 = load_a_frag(As, M + 16, kbase, hf, 264);
    acc0 = __builtin_amdgcn_wmma_f32_16x16x32_f16(false, a0, false, bf,
                                                  (short)0, acc0, false, false);
    acc1 = __builtin_amdgcn_wmma_f32_16x16x32_f16(false, a1, false, bf,
                                                  (short)0, acc1, false, false);
  }
  const float bv = bias[ncol];
#pragma unroll
  for (int v = 0; v < 8; ++v) {
    const int r0 = rowbase + v + hf * 8;
    const int r1 = r0 + 16;
    if (r0 < NROWS_O) out[(size_t)r0 * (2 * E_) + ncol] = acc0[v] + bv;
    if (r1 < NROWS_O) out[(size_t)r1 * (2 * E_) + ncol] = acc1[v] + bv;
  }
}

// K7: out[:, 256:512] = instance_feature
__global__ void k_copy_inst(const float* __restrict__ inst, float* __restrict__ out) {
  const int i = blockIdx.x * blockDim.x + threadIdx.x;
  if (i < NROWS_O * E_) {
    const int row = i >> 8, e = i & 255;
    out[(size_t)row * (2 * E_) + E_ + e] = inst[i];
  }
}

// ---------------------------------------------------------------------------
extern "C" void kernel_launch(void* const* d_in, const int* in_sizes, int n_in,
                              void* d_out, int out_size, void* d_ws, size_t ws_size,
                              hipStream_t stream) {
  const float* inst   = (const float*)d_in[0];
  const float* anchor = (const float*)d_in[1];
  const float* aemb   = (const float*)d_in[2];
  const float* proj   = (const float*)d_in[3];
  const float* iwh    = (const float*)d_in[4];
  const float* fm0    = (const float*)d_in[5];
  const float* fm1    = (const float*)d_in[6];
  const float* fm2    = (const float*)d_in[7];
  const float* fm3    = (const float*)d_in[8];
  const float* lfc_w  = (const float*)d_in[9];
  const float* lfc_b  = (const float*)d_in[10];
  const float* ce1_w  = (const float*)d_in[11];
  const float* ce1_b  = (const float*)d_in[12];
  const float* ln1_g  = (const float*)d_in[13];
  const float* ln1_b  = (const float*)d_in[14];
  const float* ce2_w  = (const float*)d_in[15];
  const float* ce2_b  = (const float*)d_in[16];
  const float* ln2_g  = (const float*)d_in[17];
  const float* ln2_b  = (const float*)d_in[18];
  const float* wfc_w  = (const float*)d_in[19];
  const float* wfc_b  = (const float*)d_in[20];
  const float* op_w   = (const float*)d_in[21];
  const float* op_b   = (const float*)d_in[22];
  float* out = (float*)d_out;

  // workspace layout: f32: ce | grid | w | fused, then f16: Bt_wfc | Bt_op
  float* ws       = (float*)d_ws;
  float* ws_ce    = ws;                                    // 12*256
  float* ws_grid  = ws_ce   + BS_ * NC_ * E_;              // 12*900*13*2
  float* ws_w     = ws_grid + BS_ * NC_ * NA_ * NPTS_ * 2; // 10800*416
  float* ws_fused = ws_w    + (size_t)NROWS_W * WOUT_;     // 1800*256
  _Float16* bt_wfc = (_Float16*)(ws_fused + (size_t)NROWS_O * E_); // 416*256
  _Float16* bt_op  = bt_wfc + (size_t)WOUT_ * E_;                  // 256*256

  k_transpose_f16<<<(E_ * WOUT_ + 255) / 256, 256, 0, stream>>>(wfc_w, bt_wfc, E_, WOUT_);
  k_transpose_f16<<<(E_ * E_    + 255) / 256, 256, 0, stream>>>(op_w,  bt_op,  E_, E_);

  k_cam_embed<<<1, 256, 0, stream>>>(proj, ce1_w, ce1_b, ln1_g, ln1_b,
                                     ce2_w, ce2_b, ln2_g, ln2_b, ws_ce);

  k_keypoints<<<(BS_ * NA_ + 255) / 256, 256, 0, stream>>>(
      inst, anchor, lfc_w, lfc_b, proj, iwh, ws_grid);

  k_wfc_gemm<<<dim3((NROWS_W + 31) / 32, (WOUT_ / 16 + 3) / 4), 128, 0, stream>>>(
      inst, aemb, ws_ce, bt_wfc, wfc_b, ws_w);

  k_softmax<<<NROWS_O, 256, 0, stream>>>(ws_w);

  k_aggregate<<<NROWS_O, 256, 0, stream>>>(fm0, fm1, fm2, fm3,
                                           ws_grid, ws_w, ws_fused);

  k_out_gemm<<<dim3((NROWS_O + 31) / 32, (E_ / 16) / 4), 128, 0, stream>>>(
      ws_fused, bt_op, op_b, out);

  k_copy_inst<<<(NROWS_O * E_ + 255) / 256, 256, 0, stream>>>(inst, out);
}
